// RGCNLayer_71133248357082
// MI455X (gfx1250) — compile-verified, matrix operand order checked by reading, not run on hardware
//
#include <hip/hip_runtime.h>

typedef __attribute__((ext_vector_type(16))) _Float16 v16h;
typedef __attribute__((ext_vector_type(8)))  _Float16 v8h;
typedef __attribute__((ext_vector_type(8)))  float    v8f;

#define IN_DIM  64
#define OUT_DIM 64
#define NUM_REL 3

// ---------------------------------------------------------------------------
// Phase 0: pack W_self + W_rel into one f16 buffer wf16[4][64][64]
// (mat 0 = self, 1..3 = relations). 32KB, done once per launch; removes all
// redundant per-wave f32->f16 weight conversions from the GEMM.
// ---------------------------------------------------------------------------
__global__ __launch_bounds__(256)
void rgcn_pack_weights(const float* __restrict__ W_self,
                       const float* __restrict__ W_rel,
                       _Float16* __restrict__ wf16) {
    int idx = blockIdx.x * blockDim.x + threadIdx.x;          // 0 .. 4*64*64-1
    const int per = OUT_DIM * IN_DIM;
    if (idx >= 4 * per) return;
    int mat = idx / per;
    int off = idx % per;
    float v = (mat == 0) ? W_self[off] : W_rel[(size_t)(mat - 1) * per + off];
    wf16[idx] = (_Float16)v;
}

// ---------------------------------------------------------------------------
// Phase 1: per-relation feature aggregation  agg[r][dst] += x[src]
// 16 lanes per edge, each lane handles 4 consecutive floats (float4 gather,
// 4x global_atomic_add_f32 scatter). x (25.6MB) and agg (76.8MB) are
// L2-resident on MI455X (192MB L2), so this runs at L2 bandwidth.
// ---------------------------------------------------------------------------
__global__ __launch_bounds__(256)
void rgcn_aggregate(const float* __restrict__ x,
                    const int* __restrict__ edge_index,
                    const int* __restrict__ edge_type,
                    float* __restrict__ agg,
                    int nEdges, int nNodes) {
    long long gid = (long long)blockIdx.x * blockDim.x + threadIdx.x;
    long long e = gid >> 4;
    if (e >= nEdges) return;
    int i = ((int)gid & 15) * 4;

    int src = edge_index[e];
    int dst = edge_index[(long long)nEdges + e];
    int t   = edge_type[e];

    const float4 v = *(const float4*)(x + (size_t)src * IN_DIM + i);
    float* p = agg + ((size_t)t * nNodes + (size_t)dst) * IN_DIM + i;

    __hip_atomic_fetch_add(p + 0, v.x, __ATOMIC_RELAXED, __HIP_MEMORY_SCOPE_AGENT);
    __hip_atomic_fetch_add(p + 1, v.y, __ATOMIC_RELAXED, __HIP_MEMORY_SCOPE_AGENT);
    __hip_atomic_fetch_add(p + 2, v.z, __ATOMIC_RELAXED, __HIP_MEMORY_SCOPE_AGENT);
    __hip_atomic_fetch_add(p + 3, v.w, __ATOMIC_RELAXED, __HIP_MEMORY_SCOPE_AGENT);
}

// ---------------------------------------------------------------------------
// f32 row (64 wide) -> two f16 WMMA operands covering K=0..31 and K=32..63.
// ISA 7.12.2 layout for 16-bit A (and mirrored B): lane half = lane>>4 selects
// K chunks {h*8 .. h*8+7} and {16+h*8 ..} per 32-K operand; v16h element
// (2*vgpr + sub) == K within those chunks.
// ---------------------------------------------------------------------------
__device__ __forceinline__
void load_row_f32_to_f16(const float* __restrict__ rowptr, int half,
                         v16h& a0, v16h& a1) {
#pragma unroll
    for (int g = 0; g < 2; ++g) {
        const int kb = g * 16 + half * 8;
#pragma unroll
        for (int j = 0; j < 8; ++j)
            a0[g * 8 + j] = (_Float16)rowptr[kb + j];
    }
#pragma unroll
    for (int g = 0; g < 2; ++g) {
        const int kb = 32 + g * 16 + half * 8;
#pragma unroll
        for (int j = 0; j < 8; ++j)
            a1[g * 8 + j] = (_Float16)rowptr[kb + j];
    }
}

// Same layout, but source is already f16: pure b128 loads, no conversions.
__device__ __forceinline__
void load_row_f16(const _Float16* __restrict__ rowptr, int half,
                  v16h& b0, v16h& b1) {
    v8h c0 = *(const v8h*)(rowptr + half * 8);          // K  h*8 .. h*8+7
    v8h c1 = *(const v8h*)(rowptr + 16 + half * 8);     // K 16+h*8 ..
    v8h c2 = *(const v8h*)(rowptr + 32 + half * 8);     // K 32+h*8 ..
    v8h c3 = *(const v8h*)(rowptr + 48 + half * 8);     // K 48+h*8 ..
    b0 = __builtin_shufflevector(c0, c1, 0,1,2,3,4,5,6,7, 8,9,10,11,12,13,14,15);
    b1 = __builtin_shufflevector(c2, c3, 0,1,2,3,4,5,6,7, 8,9,10,11,12,13,14,15);
}

// ---------------------------------------------------------------------------
// Phase 2: out = relu( x@W_selfT + sum_r agg_r@W_rT + b ), f16 WMMA, f32 acc.
// One wave computes a 16-row x 64-col output strip: 4 col-tiles x 4 source
// matrices x (K=64 -> 2 wmma) = 32 v_wmma_f32_16x16x32_f16 per wave.
// Two independent accumulator chains (K-low / K-high) double WMMA ILP;
// summed in f32 at the end, so numerics are unchanged.
// ---------------------------------------------------------------------------
__global__ __launch_bounds__(256)
void rgcn_gemm(const float* __restrict__ x,
               const float* __restrict__ agg,
               const _Float16* __restrict__ wf16,
               const float* __restrict__ b_self,
               float* __restrict__ out,
               int nNodes, int nTiles) {
    const int lane = threadIdx.x & 31;
    const int wave = threadIdx.x >> 5;
    const int tile = blockIdx.x * 8 + wave;
    if (tile >= nTiles) return;          // wave-uniform: EXEC stays all-ones
    const int row0 = tile * 16;
    const int m    = lane & 15;          // A row within tile / B output column
    const int half = lane >> 4;

    // A operands: matrix 0 = x, 1..3 = agg[r]  (loaded once, reused 4x)
    v16h A[4][2];
    load_row_f32_to_f16(x + (size_t)(row0 + m) * IN_DIM, half, A[0][0], A[0][1]);
#pragma unroll
    for (int r = 0; r < NUM_REL; ++r)
        load_row_f32_to_f16(agg + ((size_t)r * nNodes + row0 + m) * IN_DIM, half,
                            A[r + 1][0], A[r + 1][1]);

#pragma unroll
    for (int c = 0; c < 4; ++c) {
        const int n = c * 16 + m;        // output column this lane serves

        // Batch all 8 weight operands for this col-tile (one load clause).
        v16h B0[4], B1[4];
#pragma unroll
        for (int mat = 0; mat < 4; ++mat)
            load_row_f16(wf16 + ((size_t)mat * OUT_DIM + n) * IN_DIM, half,
                         B0[mat], B1[mat]);

        v8f acc0 = {};                   // K = 0..31 chain
        v8f acc1 = {};                   // K = 32..63 chain
#pragma unroll
        for (int mat = 0; mat < 4; ++mat) {
            acc0 = __builtin_amdgcn_wmma_f32_16x16x32_f16(
                       false, A[mat][0], false, B0[mat], (short)0, acc0, false, false);
            acc1 = __builtin_amdgcn_wmma_f32_16x16x32_f16(
                       false, A[mat][1], false, B1[mat], (short)0, acc1, false, false);
        }
        const float bias = b_self[n];
        // D layout: lane -> column n; VGPR v -> row (half*8 + v)
#pragma unroll
        for (int v = 0; v < 8; ++v) {
            const int row = row0 + half * 8 + v;
            float val = acc0[v] + acc1[v] + bias;
            out[(size_t)row * OUT_DIM + n] = val > 0.f ? val : 0.f;
        }
    }
}

// Scalar fallback for trailing rows (nNodes % 16) — not hit for N=100000.
__global__ void rgcn_tail(const float* __restrict__ x,
                          const float* __restrict__ agg,
                          const float* __restrict__ W_rel,
                          const float* __restrict__ W_self,
                          const float* __restrict__ b_self,
                          float* __restrict__ out,
                          int nNodes, int rowStart) {
    int idx = blockIdx.x * blockDim.x + threadIdx.x;
    int row = rowStart + idx / OUT_DIM;
    int col = idx % OUT_DIM;
    if (row >= nNodes) return;
    float s = b_self[col];
    const float* xr = x + (size_t)row * IN_DIM;
    const float* ws = W_self + (size_t)col * IN_DIM;
    for (int k = 0; k < IN_DIM; ++k) s += xr[k] * ws[k];
    for (int r = 0; r < NUM_REL; ++r) {
        const float* hr = agg + ((size_t)r * nNodes + row) * IN_DIM;
        const float* wr = W_rel + ((size_t)r * OUT_DIM + col) * IN_DIM;
        for (int k = 0; k < IN_DIM; ++k) s += hr[k] * wr[k];
    }
    out[(size_t)row * OUT_DIM + col] = s > 0.f ? s : 0.f;
}

extern "C" void kernel_launch(void* const* d_in, const int* in_sizes, int n_in,
                              void* d_out, int out_size, void* d_ws, size_t ws_size,
                              hipStream_t stream) {
    const float* x          = (const float*)d_in[0];
    const int*   edge_index = (const int*)  d_in[1];
    const int*   edge_type  = (const int*)  d_in[2];
    const float* W_rel      = (const float*)d_in[3];
    const float* W_self     = (const float*)d_in[4];
    const float* b_self     = (const float*)d_in[5];
    float*       out        = (float*)d_out;

    const int nNodes = in_sizes[0] / IN_DIM;
    const int nEdges = in_sizes[2];

    // Workspace layout: [ agg f32 : 3*N*64 ][ wf16 : 4*64*64 f16 ]
    float* agg = (float*)d_ws;
    const size_t aggBytes = (size_t)NUM_REL * nNodes * IN_DIM * sizeof(float);
    _Float16* wf16 = (_Float16*)((char*)d_ws + aggBytes);   // 16B-aligned

    hipMemsetAsync(agg, 0, aggBytes, stream);

    {   // Phase 0: one-shot f16 weight pack (4*64*64 = 16384 elements)
        const int total = 4 * OUT_DIM * IN_DIM;
        rgcn_pack_weights<<<(total + 255) / 256, 256, 0, stream>>>(
            W_self, W_rel, wf16);
    }

    {   // Phase 1: edge aggregation (16 lanes per edge)
        const long long threads = (long long)nEdges * 16;
        const int block = 256;
        const long long grid = (threads + block - 1) / block;
        rgcn_aggregate<<<(unsigned)grid, block, 0, stream>>>(
            x, edge_index, edge_type, agg, nEdges, nNodes);
    }

    {   // Phase 2: fused 4-matrix WMMA GEMM + bias + ReLU (8 waves/block)
        const int nTiles = nNodes / 16;
        if (nTiles > 0) {
            const int grid = (nTiles + 7) / 8;
            rgcn_gemm<<<grid, 256, 0, stream>>>(
                x, agg, wf16, b_self, out, nNodes, nTiles);
        }
        const int rem = nNodes - (nNodes / 16) * 16;
        if (rem > 0) {
            const int threads = rem * OUT_DIM;
            rgcn_tail<<<(threads + 255) / 256, 256, 0, stream>>>(
                x, agg, W_rel, W_self, b_self, out, nNodes, nNodes - rem);
        }
    }
}
